// RNN_KAN_29480655519756
// MI455X (gfx1250) — matrix operand order repaired and checked by reading, not compiled, not run on hardware
//
#include <hip/hip_runtime.h>
#include <hip/hip_bf16.h>
#include <math.h>

// ---------------- problem constants ----------------
#define B_    1024
#define TX    64
#define TY    16
#define TT    80            // TX + TY
#define IN0   256
#define H     512
#define GPP   9             // 8 spline coeffs + 1 silu channel per input
#define KC    (IN0 * GPP)   // 2304 combined K dimension
#define MROWS (B_ * TX)     // 65536 rows of the input-branch GEMM
#define LDA   40            // LDS row stride in halfs (padded, 16B-aligned chunks)

typedef __attribute__((ext_vector_type(16))) __bf16       v16bf;
typedef __attribute__((ext_vector_type(8)))  float        v8f;
typedef __attribute__((ext_vector_type(4)))  unsigned int u32x4;

union FragU { u32x4 q[2]; v16bf bf; };

// ---- CDNA5 async global->LDS path (ASYNCcnt), with safe fallback ----
#if __has_builtin(__builtin_amdgcn_global_load_async_to_lds_b128)
#define HAVE_ASYNC_LDS 1
typedef int v4i_vs __attribute__((vector_size(16)));
typedef __attribute__((address_space(1))) v4i_vs* g4p_t;   // global int4*
typedef __attribute__((address_space(3))) v4i_vs* l4p_t;   // LDS int4*
#else
#define HAVE_ASYNC_LDS 0
#endif

__device__ __forceinline__ void async_b128(const void* g, void* l) {
#if HAVE_ASYNC_LDS
  __builtin_amdgcn_global_load_async_to_lds_b128((g4p_t)g, (l4p_t)l, 0, 0);
#else
  *(u32x4*)l = *(const u32x4*)g;
#endif
}
__device__ __forceinline__ void async_wait() {
#if HAVE_ASYNC_LDS
#if __has_builtin(__builtin_amdgcn_s_wait_asynccnt)
  __builtin_amdgcn_s_wait_asynccnt(0);
#else
  asm volatile("s_wait_asynccnt 0" ::: "memory");
#endif
#endif
}

__device__ __forceinline__ unsigned short f2bf(float f) {
  unsigned u = __builtin_bit_cast(unsigned, f);
  u += 0x7FFFu + ((u >> 16) & 1u);          // round-to-nearest-even
  return (unsigned short)(u >> 16);
}
__device__ __forceinline__ float bf2f(unsigned short h) {
  unsigned u = ((unsigned)h) << 16;
  return __builtin_bit_cast(float, u);
}
__device__ __forceinline__ v8f vzero8() {
  v8f v = {0.f,0.f,0.f,0.f,0.f,0.f,0.f,0.f};
  return v;
}

// ---- WMMA fragment loads from LDS (CDNA5 wave32 layouts, 16-bit A 16x32 / B 32x16) ----
__device__ __forceinline__ v16bf load_fragA(const unsigned short* As, int mt, int lane) {
  int row = mt * 16 + (lane & 15);
  int kb  = (lane < 16) ? 0 : 8;            // lanes 0-15: K 0..7,16..23 ; 16-31: K 8..15,24..31
  FragU f;
  f.q[0] = *(const u32x4*)(As + row * LDA + kb);
  f.q[1] = *(const u32x4*)(As + row * LDA + 16 + kb);
  return f.bf;
}
__device__ __forceinline__ v16bf load_fragB(const unsigned short* Bst, int nt, int lane) {
  int row = nt * 16 + (lane & 15);          // B stored transposed in LDS: [N][K]
  int kb  = (lane < 16) ? 0 : 8;            // lanes 0-15: K 0..7,16..23 ; 16-31: K 8..15,24..31
  FragU f;
  f.q[0] = *(const u32x4*)(Bst + row * LDA + kb);
  f.q[1] = *(const u32x4*)(Bst + row * LDA + 16 + kb);
  return f.bf;
}

__device__ __forceinline__ void wmma_step(const unsigned short* As,
                                          const unsigned short* Bst,
                                          int wm, int wn, int lane,
                                          v8f (&acc)[2][4]) {
  v16bf a0 = load_fragA(As, wm * 2 + 0, lane);
  v16bf a1 = load_fragA(As, wm * 2 + 1, lane);
#pragma unroll
  for (int n = 0; n < 4; ++n) {
    v16bf b = load_fragB(Bst, wn * 4 + n, lane);
    acc[0][n] = __builtin_amdgcn_wmma_f32_16x16x32_bf16(false, a0, false, b,
                                                        (short)0, acc[0][n], false, false);
    acc[1][n] = __builtin_amdgcn_wmma_f32_16x16x32_bf16(false, a1, false, b,
                                                        (short)0, acc[1][n], false, false);
  }
}

// ---------------- kernel 1: pack weights to bf16, invden table, h0 copy, barrier init ----------------
__global__ void prep_kernel(const float* __restrict__ base_w,
                            const float* __restrict__ spline_w,
                            const float* __restrict__ scaler,
                            const float* __restrict__ h2h_w,
                            const float* __restrict__ grid,
                            const float* __restrict__ h0,
                            unsigned short* __restrict__ Wc,   // [H][KC] bf16, k = i*9+g
                            unsigned short* __restrict__ Wh,   // [H][H] bf16
                            float* __restrict__ invden,        // [IN0][3][12]
                            unsigned short* __restrict__ hbuf0,
                            unsigned int* __restrict__ bar) {
  long stride = (long)gridDim.x * blockDim.x;
  long tid0   = (long)blockIdx.x * blockDim.x + threadIdx.x;

  for (long idx = tid0; idx < (long)H * IN0; idx += stride) {
    int o = (int)(idx / IN0), i = (int)(idx % IN0);
    unsigned short* w = Wc + (long)o * KC + (long)i * GPP;
    float sc = scaler[idx];
    const float* sw = spline_w + idx * 8;
#pragma unroll
    for (int g = 0; g < 8; ++g) w[g] = f2bf(sw[g] * sc);
    w[8] = f2bf(base_w[idx]);                 // SiLU channel
  }
  for (long idx = tid0; idx < (long)H * H; idx += stride) Wh[idx] = f2bf(h2h_w[idx]);
  for (long i = tid0; i < IN0; i += stride) {
    const float* gr = grid + i * 12;
    float* inv = invden + i * 36;
    for (int k = 1; k <= 3; ++k)
      for (int j = 0; j < 12; ++j)
        inv[(k - 1) * 12 + j] = (j + k <= 11) ? 1.0f / (gr[j + k] - gr[j]) : 0.0f;
  }
  for (long idx = tid0; idx < (long)B_ * H; idx += stride) hbuf0[idx] = f2bf(h0[idx]);
  if (tid0 < 16) bar[tid0] = 0u;
}

// ---------------- kernel 2: decoder constant z = kan_linear(0) ----------------
__global__ void zvec_kernel(const float* __restrict__ grid,
                            const float* __restrict__ invden,
                            const float* __restrict__ spline_w,
                            const float* __restrict__ scaler,
                            float* __restrict__ z) {
  int o = blockIdx.x * blockDim.x + threadIdx.x;
  if (o >= H) return;
  const float* gr  = grid;     // all grid rows identical
  const float* inv = invden;
  float bb[11];
  const float xv = 0.0f;
#pragma unroll
  for (int j = 0; j < 11; ++j) bb[j] = (xv >= gr[j] && xv < gr[j + 1]) ? 1.f : 0.f;
#pragma unroll
  for (int j = 0; j < 10; ++j) bb[j] = (xv - gr[j]) * inv[j]      * bb[j] + (gr[j + 2] - xv) * inv[j + 1]      * bb[j + 1];
#pragma unroll
  for (int j = 0; j < 9; ++j)  bb[j] = (xv - gr[j]) * inv[12 + j] * bb[j] + (gr[j + 3] - xv) * inv[12 + j + 1] * bb[j + 1];
#pragma unroll
  for (int j = 0; j < 8; ++j)  bb[j] = (xv - gr[j]) * inv[24 + j] * bb[j] + (gr[j + 4] - xv) * inv[24 + j + 1] * bb[j + 1];
  float acc = 0.f;
  for (int i = 0; i < IN0; ++i) {
    const float* sw = spline_w + ((long)o * IN0 + i) * 8;
    float sc = scaler[(long)o * IN0 + i];
    float s = 0.f;
#pragma unroll
    for (int g = 0; g < 8; ++g) s += bb[g] * sw[g];
    acc += s * sc;                              // silu(0)=0 -> base branch is zero
  }
  z[o] = acc;
}

// ---------------- kernel 3: fused KAN input-branch GEMM  U = Act(x) @ Wc^T ----------------
// block tile M=64 x N=256, 8 waves (2x4), bases generated on the fly into LDS
__global__ __launch_bounds__(256) void kan_gemm_kernel(const float* __restrict__ x,
                                                       const float* __restrict__ grid,
                                                       const float* __restrict__ invden,
                                                       const unsigned short* __restrict__ Wc,
                                                       float* __restrict__ U) {
  extern __shared__ unsigned char smem[];
  unsigned short* Xs  = (unsigned short*)smem;                   // 64*256 bf16 = 32768 B
  unsigned short* As  = (unsigned short*)(smem + 32768);         // 64*LDA   =  5120 B
  unsigned short* Bst = (unsigned short*)(smem + 32768 + 5120);  // 256*LDA  = 20480 B

  const int tid  = threadIdx.x;
  const int lane = tid & 31;
  const int wave = tid >> 5;
  const int wm = wave >> 2, wn = wave & 3;
  const long m0 = (long)blockIdx.x * 64;
  const int  n0 = blockIdx.y * 256;

  // stage x tile (64 rows x 256 inputs) -> LDS bf16
  for (int idx = tid; idx < 64 * IN0; idx += 256) {
    int r = idx >> 8, i = idx & 255;
    Xs[idx] = f2bf(x[(m0 + r) * IN0 + i]);
  }

  v8f acc[2][4];
#pragma unroll
  for (int mi = 0; mi < 2; ++mi)
#pragma unroll
    for (int n = 0; n < 4; ++n) acc[mi][n] = vzero8();

  const int NKT = KC / 32;   // 72 K-tiles
  for (int kt = 0; kt < NKT; ++kt) {
    const int k0 = kt * 32;
    __syncthreads();
    // weight tile: thread n async-copies one K-slice row of Wc (2 x b128, 16B aligned)
    {
      const unsigned short* src = Wc + (long)(n0 + tid) * KC + k0;
      unsigned short* dst = Bst + tid * LDA;
      async_b128(src, dst);
      async_b128(src + 8, dst + 8);
      if (kt + 1 < NKT) __builtin_prefetch(src + 32, 0, 3);
    }
    // activation tile: compute bases + silu for the inputs spanning [k0, k0+32)
    {
      const int i_first = k0 / GPP;       // constant divide per tile
      for (int w = tid; w < 64 * 5; w += 256) {   // fixed 5 inputs/row (mask extras)
        int r = w / 5;                     // constant-divisor -> magic multiply
        int i = i_first + (w - r * 5);
        if (i >= IN0) continue;
        float xv = bf2f(Xs[(r << 8) + i]);
        const float* gr  = grid + i * 12;
        const float* inv = invden + i * 36;
        float bb[11];
#pragma unroll
        for (int j = 0; j < 11; ++j) bb[j] = (xv >= gr[j] && xv < gr[j + 1]) ? 1.f : 0.f;
#pragma unroll
        for (int j = 0; j < 10; ++j) bb[j] = (xv - gr[j]) * inv[j]      * bb[j] + (gr[j + 2] - xv) * inv[j + 1]      * bb[j + 1];
#pragma unroll
        for (int j = 0; j < 9; ++j)  bb[j] = (xv - gr[j]) * inv[12 + j] * bb[j] + (gr[j + 3] - xv) * inv[12 + j + 1] * bb[j + 1];
#pragma unroll
        for (int j = 0; j < 8; ++j)  bb[j] = (xv - gr[j]) * inv[24 + j] * bb[j] + (gr[j + 4] - xv) * inv[24 + j + 1] * bb[j + 1];
        float silu = xv / (1.0f + __expf(-xv));
        int kbase = i * GPP - k0;
#pragma unroll
        for (int g = 0; g < 8; ++g) {
          int kk = kbase + g;
          if ((unsigned)kk < 32u) As[r * LDA + kk] = f2bf(bb[g]);
        }
        int kk = kbase + 8;
        if ((unsigned)kk < 32u) As[r * LDA + kk] = f2bf(silu);
      }
    }
    async_wait();
    __syncthreads();
    wmma_step(As, Bst, wm, wn, lane, acc);
  }

  // epilogue: write fp32 U (layout per 16x16 f32 C/D fragment)
  const int rsel = (lane < 16) ? 0 : 8;
  const int csel = lane & 15;
#pragma unroll
  for (int mi = 0; mi < 2; ++mi)
#pragma unroll
    for (int n = 0; n < 4; ++n) {
      long gm = m0 + (wm * 2 + mi) * 16;
      int  gn = n0 + (wn * 4 + n) * 16 + csel;
#pragma unroll
      for (int r = 0; r < 8; ++r)
        U[(gm + rsel + r) * H + gn] = acc[mi][n][r];
    }
}

// ---------------- kernel 4: persistent recurrence, 80 steps with grid-wide barrier ----------------
// grid = (16, 2) = 32 co-resident blocks; tile M=64 x N=256, K=512; h ping-pong kept in bf16
__global__ __launch_bounds__(256) void rnn_kernel(const unsigned short* __restrict__ Wh,
                                                  const float* __restrict__ U,
                                                  const float* __restrict__ z,
                                                  const float* __restrict__ h2h_b,
                                                  unsigned short* __restrict__ hb0,
                                                  unsigned short* __restrict__ hb1,
                                                  float* __restrict__ h_all,
                                                  unsigned int* __restrict__ bar) {
  extern __shared__ unsigned char smem[];
  unsigned short* As  = (unsigned short*)smem;            // 64*LDA  =  5120 B
  unsigned short* Bst = (unsigned short*)(smem + 5120);   // 256*LDA = 20480 B

  const int tid  = threadIdx.x;
  const int lane = tid & 31;
  const int wave = tid >> 5;
  const int wm = wave >> 2, wn = wave & 3;
  const int m0 = blockIdx.x * 64;                         // batch rows
  const int n0 = blockIdx.y * 256;
  const unsigned nBlocks = gridDim.x * gridDim.y;         // 32

  for (int t = 0; t < TT; ++t) {
    const unsigned short* hcur = (t & 1) ? hb1 : hb0;
    unsigned short*       hnxt = (t & 1) ? hb0 : hb1;

    v8f acc[2][4];
#pragma unroll
    for (int mi = 0; mi < 2; ++mi)
#pragma unroll
      for (int n = 0; n < 4; ++n) acc[mi][n] = vzero8();

    for (int kt = 0; kt < H / 32; ++kt) {
      __syncthreads();
      {  // weight tile: async b128 pairs
        const unsigned short* src = Wh + (long)(n0 + tid) * H + kt * 32;
        unsigned short* dst = Bst + tid * LDA;
        async_b128(src, dst);
        async_b128(src + 8, dst + 8);
      }
      {  // h tile: pure async copy, bf16 already (thread -> row tid/4, chunk tid%4)
        int r = tid >> 2, c = tid & 3;
        async_b128(hcur + (long)(m0 + r) * H + kt * 32 + c * 8,
                   As + r * LDA + c * 8);
      }
      async_wait();
      __syncthreads();
      wmma_step(As, Bst, wm, wn, lane, acc);
    }

    // epilogue: u_t (encoder: precomputed KAN branch, decoder: constant z), tanh, store
    const int rsel = (lane < 16) ? 0 : 8;
    const int csel = lane & 15;
#pragma unroll
    for (int mi = 0; mi < 2; ++mi)
#pragma unroll
      for (int n = 0; n < 4; ++n) {
        int gb = m0 + (wm * 2 + mi) * 16;
        int gn = n0 + (wn * 4 + n) * 16 + csel;
        float bias = h2h_b[gn];
#pragma unroll
        for (int r = 0; r < 8; ++r) {
          int b = gb + rsel + r;
          float u = (t < TX) ? U[((long)b * TX + t) * H + gn] : z[gn];
          float hv = tanhf(acc[mi][n][r] + u + bias);
          hnxt[(long)b * H + gn] = f2bf(hv);
          h_all[((long)b * TT + t) * H + gn] = hv;
        }
      }

    // device-wide barrier (monotone counter: no reset needed within the launch)
    __syncthreads();
    if (tid == 0) {
      __threadfence();
      atomicAdd(bar, 1u);
      const unsigned target = nBlocks * (unsigned)(t + 1);
      while (__hip_atomic_load(bar, __ATOMIC_ACQUIRE, __HIP_MEMORY_SCOPE_AGENT) < target) {
        __builtin_amdgcn_s_sleep(1);
      }
    }
    __syncthreads();
  }
}

// ---------------- kernel 5: y_pred head ----------------
__global__ void head_kernel(const float* __restrict__ h_all,
                            const float* __restrict__ fc_w,
                            const float* __restrict__ fc_b,
                            float* __restrict__ y) {
  int idx = blockIdx.x * blockDim.x + threadIdx.x;
  if (idx >= B_ * TY) return;
  int b = idx >> 4, t = idx & 15;
  const float* hp = h_all + ((long)b * TT + TX + t) * H;
  float acc = fc_b[0];
  for (int j = 0; j < H; j += 4)
    acc += hp[j] * fc_w[j] + hp[j + 1] * fc_w[j + 1] + hp[j + 2] * fc_w[j + 2] + hp[j + 3] * fc_w[j + 3];
  y[idx] = acc;
}

// ---------------- host-side orchestration ----------------
extern "C" void kernel_launch(void* const* d_in, const int* in_sizes, int n_in,
                              void* d_out, int out_size, void* d_ws, size_t ws_size,
                              hipStream_t stream) {
  (void)in_sizes; (void)n_in; (void)out_size; (void)ws_size;

  const float* x        = (const float*)d_in[0];
  const float* h0       = (const float*)d_in[1];
  const float* grid     = (const float*)d_in[2];
  const float* base_w   = (const float*)d_in[3];
  const float* spline_w = (const float*)d_in[4];
  const float* scaler   = (const float*)d_in[5];
  const float* h2h_w    = (const float*)d_in[6];
  const float* h2h_b    = (const float*)d_in[7];
  const float* fc_w     = (const float*)d_in[8];
  const float* fc_b     = (const float*)d_in[9];

  float* y_out = (float*)d_out;                      // (B, TY)
  float* h_all = y_out + (size_t)B_ * TY;            // (B, TT, H)

  unsigned char* ws = (unsigned char*)d_ws;
  // workspace layout (all 256B-aligned), total ~139 MB
  const size_t oWc  = 0;                                     // H*KC bf16     = 2,359,296 B
  const size_t oWh  = oWc  + (size_t)H * KC * 2;             // H*H bf16      =   524,288 B
  const size_t oInv = oWh  + (size_t)H * H * 2;              // IN0*36 f32    =    36,864 B
  const size_t oZ   = oInv + (size_t)IN0 * 36 * 4;           // H f32         =     2,048 B
  const size_t oBar = oZ   + (size_t)H * 4;                  // counters      =       256 B
  const size_t oHb0 = oBar + 256;                            // B*H bf16      = 1,048,576 B
  const size_t oHb1 = oHb0 + (size_t)B_ * H * 2;
  const size_t oU   = oHb1 + (size_t)B_ * H * 2;             // MROWS*H f32   = 134,217,728 B

  unsigned short* Wc   = (unsigned short*)(ws + oWc);
  unsigned short* Wh   = (unsigned short*)(ws + oWh);
  float*        invden = (float*)(ws + oInv);
  float*        zvec   = (float*)(ws + oZ);
  unsigned int* bar    = (unsigned int*)(ws + oBar);
  unsigned short* hb0  = (unsigned short*)(ws + oHb0);
  unsigned short* hb1  = (unsigned short*)(ws + oHb1);
  float*        U      = (float*)(ws + oU);

  prep_kernel<<<512, 256, 0, stream>>>(base_w, spline_w, scaler, h2h_w, grid, h0,
                                       Wc, Wh, invden, hb0, bar);
  zvec_kernel<<<2, 256, 0, stream>>>(grid, invden, spline_w, scaler, zvec);

  dim3 g1(MROWS / 64, 2);
  kan_gemm_kernel<<<g1, 256, 32768 + 5120 + 20480, stream>>>(x, grid, invden, Wc, U);

  dim3 g2(B_ / 64, 2);   // 32 co-resident blocks for the persistent recurrence
  rnn_kernel<<<g2, 256, 5120 + 20480, stream>>>(Wh, U, zvec, h2h_b, hb0, hb1, h_all, bar);

  head_kernel<<<(B_ * TY + 255) / 256, 256, 0, stream>>>(h_all, fc_w, fc_b, y_out);
}
